// SchnetLayer_59399397703798
// MI455X (gfx1250) — matrix operand order repaired and checked by reading, not compiled, not run on hardware
//
#include <hip/hip_runtime.h>
#include <hip/hip_bf16.h>

typedef __attribute__((ext_vector_type(16))) _Float16 v16h;
typedef __attribute__((ext_vector_type(8)))  _Float16 v8h;
typedef __attribute__((ext_vector_type(8)))  float    v8f;
typedef __attribute__((ext_vector_type(4)))  unsigned int u32x4;
typedef __attribute__((ext_vector_type(8)))  int i32x8;
typedef __attribute__((ext_vector_type(4)))  int i32x4;

#define LN2F 0.69314718055994531f
#define PIF  3.14159265358979323f

__device__ __forceinline__ v8f wmma32(v16h a, v16h b, v8f c) {
  return __builtin_amdgcn_wmma_f32_16x16x32_f16(false, a, false, b, (short)0, c, false, false);
}

__device__ __forceinline__ v16h cat8(v8h lo, v8h hi) {
  return __builtin_shufflevector(lo, hi, 0,1,2,3,4,5,6,7,8,9,10,11,12,13,14,15);
}

// A fragment: 16x32 f16 tile, row-major source with leading dim K (elements).
__device__ __forceinline__ v16h load_a_frag(const _Float16* A, int m, int K, int kb, int half) {
  const _Float16* p = A + (size_t)m * K + kb * 32 + half * 8;
  v8h lo = *(const v8h*)p;
  v8h hi = *(const v8h*)(p + 16);
  return cat8(lo, hi);
}

// B fragment from pre-packed layout: 16 contiguous halfs per lane (32 bytes).
__device__ __forceinline__ v16h load_b_packed(const _Float16* Bp, int kb, int nt, int ntiles, int lane) {
  return *(const v16h*)(Bp + (((size_t)(kb * ntiles + nt) * 32 + lane) * 16));
}

__device__ __forceinline__ float sspf(float x) {
  return fmaxf(x, 0.f) + __logf(1.f + __expf(-fabsf(x))) - LN2F;
}

// TDM bulk copy global -> LDS (1-D tile, 8-byte elements). bytes % 8 == 0,
// bytes/8 < 65536. Issued once per calling wave; tracked by TENSORcnt.
__device__ __forceinline__ void tdm_load(unsigned lds_off, const void* gsrc, unsigned bytes) {
  unsigned long long ga = (unsigned long long)gsrc;
  unsigned n8 = bytes >> 3;
  u32x4 g0;
  g0[0] = 1u;                                                  // count=1 (user mode)
  g0[1] = lds_off;                                             // D#.lds_addr
  g0[2] = (unsigned)(ga & 0xFFFFFFFFu);                        // global_addr[31:0]
  g0[3] = (unsigned)((ga >> 32) & 0x01FFFFFFu) | 0x80000000u;  // [56:32] | type=2
  i32x8 g1;
  g1[0] = (int)(3u << 16);                 // wg_mask=0, data_size=3 (8B)
  g1[1] = (int)((n8 & 0xFFFFu) << 16);     // tensor_dim0[15:0]
  g1[2] = (int)(((n8 >> 16) & 0xFFFFu) | (1u << 16));  // tensor_dim0[31:16], tensor_dim1=1
  g1[3] = (int)((n8 & 0xFFFFu) << 16);     // tile_dim0
  g1[4] = 0;                               // tile_dim1=0 (1-D), tile_dim2=0
  g1[5] = (int)n8;                         // tensor_dim0_stride[31:0]
  g1[6] = 0;
  g1[7] = 0;
  i32x4 gz = {};
  i32x8 gz8 = {};
  __builtin_amdgcn_tensor_load_to_lds(g0, g1, gz, gz, gz8, 0);
}

// ---------------------------------------------------------------------------
// utility kernels
// ---------------------------------------------------------------------------
__global__ void k_zero(float* p, long n) {
  long i = (long)blockIdx.x * blockDim.x + threadIdx.x;
  if (i < n) p[i] = 0.f;
}

__global__ void k_pad16(const float* src, _Float16* dst, int rows, int creal, int cpad) {
  long i = (long)blockIdx.x * blockDim.x + threadIdx.x;
  long total = (long)rows * cpad;
  if (i >= total) return;
  int r = (int)(i / cpad), c = (int)(i % cpad);
  float v = (c < creal) ? src[(size_t)r * creal + c] : 0.f;
  dst[i] = (_Float16)v;
}

// pack f32 weight [Kreal, Nreal] row-major into WMMA B-fragment order (f16).
__global__ void k_pack_b(const float* W, _Float16* out, int Kreal, int Nreal, int Kpad, int Npad) {
  int ntiles = Npad >> 4;
  int total = (Kpad >> 5) * ntiles * 32;
  int t = blockIdx.x * blockDim.x + threadIdx.x;
  if (t >= total) return;
  int lane = t & 31;
  int tl = t >> 5;
  int kb = tl / ntiles, nt = tl % ntiles;
  int half = lane >> 4, nl = lane & 15;
  int n = nt * 16 + nl;
  _Float16* o = out + ((size_t)(kb * ntiles + nt) * 32 + lane) * 16;
#pragma unroll
  for (int j = 0; j < 16; ++j) {
    int k = kb * 32 + ((j >> 3) << 4) + half * 8 + (j & 7);
    float v = (k < Kreal && n < Nreal) ? W[(size_t)k * Nreal + n] : 0.f;
    o[j] = (_Float16)v;
  }
}

__global__ void k_edge_geom(const float* pos, const int* ei, float* dist, float* cosC, int E) {
  long e = (long)blockIdx.x * blockDim.x + threadIdx.x;
  if (e >= E) return;
  int j = ei[e], i2 = ei[(long)E + e];
  float dx = pos[(size_t)j * 3 + 0] - pos[(size_t)i2 * 3 + 0];
  float dy = pos[(size_t)j * 3 + 1] - pos[(size_t)i2 * 3 + 1];
  float dz = pos[(size_t)j * 3 + 2] - pos[(size_t)i2 * 3 + 2];
  float d = sqrtf(dx * dx + dy * dy + dz * dz);
  dist[e] = d;
  cosC[e] = 0.5f * (__cosf(d * (PIF / 6.0f)) + 1.0f);
}

__global__ void k_bn_stats(const float* h, float* stats, int Nn) {
  int c = blockIdx.x;
  float s = 0.f, sq = 0.f;
  for (int n = threadIdx.x; n < Nn; n += blockDim.x) {
    float x = h[(size_t)n * 64 + c];
    s += x; sq += x * x;
  }
  __shared__ float bs[256], bq[256];
  bs[threadIdx.x] = s; bq[threadIdx.x] = sq;
  __syncthreads();
  for (int o = 128; o > 0; o >>= 1) {
    if ((int)threadIdx.x < o) { bs[threadIdx.x] += bs[threadIdx.x + o]; bq[threadIdx.x] += bq[threadIdx.x + o]; }
    __syncthreads();
  }
  if (threadIdx.x == 0) { stats[c] = bs[0]; stats[64 + c] = bq[0]; }
}

__global__ void k_bn_apply(const float* h, const float* stats, const float* gamma,
                           const float* beta, _Float16* out, int Nn) {
  long i = (long)blockIdx.x * blockDim.x + threadIdx.x;
  if (i >= (long)Nn * 64) return;
  int c = (int)(i & 63);
  float mu = stats[c] / (float)Nn;
  float var = stats[64 + c] / (float)Nn - mu * mu;
  float sc = gamma[c] * rsqrtf(var + 1e-5f);
  float x = (h[i] - mu) * sc + beta[c];
  out[i] = (_Float16)fmaxf(x, 0.f);
}

__global__ void k_scatter_out(const float* s2, const int* batch, float* out, int Nn) {
  long i = (long)blockIdx.x * blockDim.x + threadIdx.x;
  if (i >= (long)Nn * 32) return;
  int n = (int)(i >> 5), c = (int)(i & 31);
  atomicAdd(&out[(size_t)batch[n] * 32 + c], s2[i]);
}

// ---------------------------------------------------------------------------
// WMMA GEMM, one wave computes a 16 x (16*NT) output strip with NT independent
// accumulators. act: 0=none 1=ssp 2=relu
// ---------------------------------------------------------------------------
template <int NT>
__global__ void k_gemm_t(const _Float16* __restrict__ A, const _Float16* __restrict__ Bp,
                         const float* __restrict__ bias, int M, int K, int Nc, int act,
                         _Float16* __restrict__ out16, float* __restrict__ out32,
                         const float* __restrict__ resid) {
  int lane = threadIdx.x & 31;
  int wave = threadIdx.x >> 5;
  int ntiles = Nc >> 4;
  int ngroups = ntiles / NT;
  long tile = (long)blockIdx.x * (blockDim.x >> 5) + wave;
  long total = (long)(M >> 4) * ngroups;
  if (tile >= total) return;
  int tm = (int)(tile / ngroups), tg = (int)(tile % ngroups);
  int half = lane >> 4, nl = lane & 15;

  v8f zero = {};
  v8f acc[NT];
#pragma unroll
  for (int j = 0; j < NT; ++j) acc[j] = zero;

  int ksteps = K >> 5;
  for (int kb = 0; kb < ksteps; ++kb) {
    v16h a = load_a_frag(A, tm * 16 + nl, K, kb, half);
#pragma unroll
    for (int j = 0; j < NT; ++j) {
      v16h b = load_b_packed(Bp, kb, tg * NT + j, ntiles, lane);
      acc[j] = wmma32(a, b, acc[j]);
    }
  }
#pragma unroll
  for (int j = 0; j < NT; ++j) {
    int tn = tg * NT + j;
    float bi = bias ? bias[tn * 16 + nl] : 0.f;
#pragma unroll
    for (int r = 0; r < 8; ++r) {
      float x = acc[j][r] + bi;
      if (act == 1) x = sspf(x);
      else if (act == 2) x = fmaxf(x, 0.f);
      size_t idx = (size_t)(tm * 16 + half * 8 + r) * Nc + tn * 16 + nl;
      if (resid) x += resid[idx];
      if (out32) out32[idx] = x;
      if (out16) out16[idx] = (_Float16)x;
    }
  }
}

// ---------------------------------------------------------------------------
// fused edge kernel:
//   W = (ssp(gauss(dist) @ m1 + b1) @ m2 + b2) * cosC ;  agg[col] += W * vlin[row]
// one wave per 16-edge tile. mlp_w2 (32KB) staged to LDS once per block via the
// Tensor Data Mover; ssp intermediate and gathered vlin rows staged through
// per-wave LDS; n-tiles in pairs (independent accumulators).
// ---------------------------------------------------------------------------
__global__ void k_edge(const float* __restrict__ dist, const float* __restrict__ cosC,
                       const int* __restrict__ rowi, const int* __restrict__ coli,
                       const _Float16* __restrict__ m1p, const float* __restrict__ b1,
                       const _Float16* __restrict__ m2p, const float* __restrict__ b2,
                       const _Float16* __restrict__ vlin, float* __restrict__ agg, int E) {
  __shared__ __align__(32) _Float16 sw2[128 * 128];    // 32KB: staged mlp_w2 fragments
  __shared__ __align__(32) _Float16 lds[8][16 * 128];  // 4KB per wave
  int lane = threadIdx.x & 31;
  int wave = threadIdx.x >> 5;

  // ---- TDM: stage packed mlp_w2 into LDS (one copy per workgroup) ----
  if (threadIdx.x < 32) {
    unsigned w2off = (unsigned)(size_t)(&sw2[0]);  // flat LDS aperture: low 32 bits = offset
    tdm_load(w2off, m2p, 128 * 128 * 2);
    __builtin_amdgcn_s_wait_tensorcnt((short)0);
  }
  __syncthreads();

  long e0 = ((long)blockIdx.x * 8 + wave) * 16;
  if (e0 >= E) return;
  int half = lane >> 4, nl = lane & 15;

  const float step = 6.0f / 49.0f;
  const float coeff = -0.5f / (step * step);

  // Gaussian basis A-fragments (K padded 50 -> 64)
  float d = dist[e0 + nl];
  v16h a1[2];
#pragma unroll
  for (int kb = 0; kb < 2; ++kb) {
#pragma unroll
    for (int i = 0; i < 16; ++i) {
      int k = kb * 32 + ((i >> 3) << 4) + half * 8 + (i & 7);
      float dk = d - (float)k * step;
      float v = (k < 50) ? __expf(coeff * dk * dk) : 0.f;
      a1[kb][i] = (_Float16)v;
    }
  }

  // GEMM1: [16,64] @ [64,128] -> ssp -> LDS, n-tiles in pairs
  _Float16* hl = lds[wave];
#pragma unroll
  for (int nt = 0; nt < 8; nt += 2) {
    v8f acc0 = {}, acc1 = {};
#pragma unroll
    for (int kb = 0; kb < 2; ++kb) {
      v16h b0 = load_b_packed(m1p, kb, nt, 8, lane);
      v16h b1f = load_b_packed(m1p, kb, nt + 1, 8, lane);
      acc0 = wmma32(a1[kb], b0, acc0);
      acc1 = wmma32(a1[kb], b1f, acc1);
    }
    float bi0 = b1[nt * 16 + nl];
    float bi1 = b1[(nt + 1) * 16 + nl];
#pragma unroll
    for (int r = 0; r < 8; ++r) {
      hl[(half * 8 + r) * 128 + nt * 16 + nl]       = (_Float16)sspf(acc0[r] + bi0);
      hl[(half * 8 + r) * 128 + (nt + 1) * 16 + nl] = (_Float16)sspf(acc1[r] + bi1);
    }
  }
  asm volatile("s_wait_dscnt 0" ::: "memory");  // cross-lane LDS RAW (same wave)

  // reload ssp intermediate as A-fragments (transpose via LDS)
  v16h a2[4];
#pragma unroll
  for (int kb = 0; kb < 4; ++kb) {
    const _Float16* p = hl + nl * 128 + kb * 32 + half * 8;
    v8h lo = *(const v8h*)p;
    v8h hi = *(const v8h*)(p + 16);
    a2[kb] = cat8(lo, hi);
  }
  asm volatile("s_wait_dscnt 0" ::: "memory");  // a2 reads done before hl reuse

  // stage gathered vlin rows into LDS with 16B vector loads (2 lanes/row)
  {
    int rr = lane >> 1, hh = lane & 1;
    int srow = rowi[e0 + rr];
    const _Float16* vsrc = vlin + (size_t)srow * 128 + hh * 64;
    _Float16* ldst = hl + rr * 128 + hh * 64;
#pragma unroll
    for (int j = 0; j < 8; ++j)
      *(v8h*)(ldst + j * 8) = *(const v8h*)(vsrc + j * 8);
  }

  // per-lane edge metadata for its 8 output rows
  float cc[8]; int dst[8];
#pragma unroll
  for (int r = 0; r < 8; ++r) {
    long er = e0 + half * 8 + r;
    cc[r] = cosC[er];
    dst[r] = coli[er];
  }
  asm volatile("s_wait_dscnt 0" ::: "memory");  // vlin tile visible in LDS

  // GEMM2: [16,128] @ [128,128] with B from LDS; gate, multiply, scatter-add
#pragma unroll
  for (int nt = 0; nt < 8; nt += 2) {
    v8f acc0 = {}, acc1 = {};
#pragma unroll
    for (int kb = 0; kb < 4; ++kb) {
      v16h b0 = *(const v16h*)(sw2 + (((kb * 8 + nt) * 32 + lane) * 16));
      v16h b1f = *(const v16h*)(sw2 + (((kb * 8 + nt + 1) * 32 + lane) * 16));
      acc0 = wmma32(a2[kb], b0, acc0);
      acc1 = wmma32(a2[kb], b1f, acc1);
    }
    float bi0 = b2[nt * 16 + nl];
    float bi1 = b2[(nt + 1) * 16 + nl];
    int ng0 = nt * 16 + nl;
    int ng1 = (nt + 1) * 16 + nl;
#pragma unroll
    for (int r = 0; r < 8; ++r) {
      int rl = half * 8 + r;
      float w0 = (acc0[r] + bi0) * cc[r];
      float w1 = (acc1[r] + bi1) * cc[r];
      float v0 = w0 * (float)hl[rl * 128 + ng0];
      float v1 = w1 * (float)hl[rl * 128 + ng1];
      atomicAdd(&agg[(size_t)dst[r] * 128 + ng0], v0);
      atomicAdd(&agg[(size_t)dst[r] * 128 + ng1], v1);
    }
  }
}

// ---------------------------------------------------------------------------
extern "C" void kernel_launch(void* const* d_in, const int* in_sizes, int n_in,
                              void* d_out, int out_size, void* d_ws, size_t ws_size,
                              hipStream_t stream) {
  const float* x       = (const float*)d_in[0];
  const float* pos     = (const float*)d_in[1];
  const int*   batch   = (const int*)d_in[2];
  const int*   ei      = (const int*)d_in[3];
  const float* fe_w1   = (const float*)d_in[4];
  const float* fe_b1   = (const float*)d_in[5];
  const float* bn_g    = (const float*)d_in[6];
  const float* bn_b    = (const float*)d_in[7];
  const float* fe_w2   = (const float*)d_in[8];
  const float* fe_b2   = (const float*)d_in[9];
  const float* lin_w   = (const float*)d_in[10];
  const float* mlp_w1  = (const float*)d_in[11];
  const float* mlp_b1  = (const float*)d_in[12];
  const float* mlp_w2  = (const float*)d_in[13];
  const float* mlp_b2  = (const float*)d_in[14];
  const float* v1_w    = (const float*)d_in[15];
  const float* v1_b    = (const float*)d_in[16];
  const float* v2_w    = (const float*)d_in[17];
  const float* v2_b    = (const float*)d_in[18];
  const float* u1_w    = (const float*)d_in[19];
  const float* u1_b    = (const float*)d_in[20];
  const float* u2_w    = (const float*)d_in[21];
  const float* u2_b    = (const float*)d_in[22];

  const int Nn = in_sizes[0] / 28;            // 50000
  const int E  = in_sizes[3] / 2;             // 800000
  const int L  = in_sizes[10] / (128 * 128);  // 6
  float* out = (float*)d_out;

  // ---- workspace carve-up (256B aligned) ----
  char* ws = (char*)d_ws;
  size_t off = 0;
  auto take = [&](size_t bytes) -> char* {
    off = (off + 255) & ~(size_t)255;
    char* p = ws + off; off += bytes; return p;
  };
  _Float16* xh   = (_Float16*)take((size_t)Nn * 32 * 2);
  _Float16* vh   = (_Float16*)take((size_t)Nn * 128 * 2);
  float*    vf   = (float*)   take((size_t)Nn * 128 * 4);
  _Float16* sA   = (_Float16*)take((size_t)Nn * 128 * 2);  // h16 / vlin / aggh
  _Float16* sB   = (_Float16*)take((size_t)Nn * 128 * 2);  // t
  float*    agg  = (float*)   take((size_t)Nn * 128 * 4);  // also hbuf / s2
  float*    dist = (float*)   take((size_t)E * 4);
  float*    cosC = (float*)   take((size_t)E * 4);
  float*    stats= (float*)   take(128 * 4);
  _Float16* pw_fe1 = (_Float16*)take(32 * 64 * 2);
  _Float16* pw_fe2 = (_Float16*)take(64 * 128 * 2);
  _Float16* pw_u1  = (_Float16*)take(128 * 64 * 2);
  _Float16* pw_u2  = (_Float16*)take(64 * 32 * 2);
  _Float16 *pw_lin[8], *pw_m1[8], *pw_m2[8], *pw_v1[8], *pw_v2[8];
  for (int l = 0; l < L; ++l) {
    pw_lin[l] = (_Float16*)take(128 * 128 * 2);
    pw_m1[l]  = (_Float16*)take(64 * 128 * 2);
    pw_m2[l]  = (_Float16*)take(128 * 128 * 2);
    pw_v1[l]  = (_Float16*)take(128 * 128 * 2);
    pw_v2[l]  = (_Float16*)take(128 * 128 * 2);
  }
  (void)ws_size; (void)n_in;

  auto pack = [&](const float* W, _Float16* dstp, int Kr, int Nr, int Kp, int Np) {
    int threads = (Kp / 32) * (Np / 16) * 32;
    k_pack_b<<<(threads + 255) / 256, 256, 0, stream>>>(W, dstp, Kr, Nr, Kp, Np);
  };
  auto gemm = [&](const _Float16* A, const _Float16* Bp, const float* bias,
                  int M, int K, int Nc, int act, _Float16* o16, float* o32,
                  const float* resid) {
    if (Nc >= 64) {
      long tiles = (long)(M / 16) * (Nc / 64);
      unsigned blocks = (unsigned)((tiles + 7) / 8);
      k_gemm_t<4><<<blocks, 256, 0, stream>>>(A, Bp, bias, M, K, Nc, act, o16, o32, resid);
    } else {
      long tiles = (long)(M / 16) * (Nc / 32);
      unsigned blocks = (unsigned)((tiles + 7) / 8);
      k_gemm_t<2><<<blocks, 256, 0, stream>>>(A, Bp, bias, M, K, Nc, act, o16, o32, resid);
    }
  };
  auto zero = [&](float* p, long n) {
    k_zero<<<(unsigned)((n + 255) / 256), 256, 0, stream>>>(p, n);
  };

  // ---- pack all weights to f16 B-fragment layout ----
  pack(fe_w1, pw_fe1, 28, 64, 32, 64);
  pack(fe_w2, pw_fe2, 64, 128, 64, 128);
  pack(u1_w,  pw_u1, 128, 64, 128, 64);
  pack(u2_w,  pw_u2, 64, 32, 64, 32);
  for (int l = 0; l < L; ++l) {
    pack(lin_w  + (size_t)l * 128 * 128, pw_lin[l], 128, 128, 128, 128);
    pack(mlp_w1 + (size_t)l * 50 * 128,  pw_m1[l],  50, 128, 64, 128);
    pack(mlp_w2 + (size_t)l * 128 * 128, pw_m2[l], 128, 128, 128, 128);
    pack(v1_w   + (size_t)l * 128 * 128, pw_v1[l], 128, 128, 128, 128);
    pack(v2_w   + (size_t)l * 128 * 128, pw_v2[l], 128, 128, 128, 128);
  }

  // ---- feature embedding ----
  {
    long t = (long)Nn * 32;
    k_pad16<<<(unsigned)((t + 255) / 256), 256, 0, stream>>>(x, xh, Nn, 28, 32);
  }
  float* hbuf = agg;  // reuse [Nn,64] before layers start
  gemm(xh, pw_fe1, fe_b1, Nn, 32, 64, /*act*/0, nullptr, hbuf, nullptr);
  k_bn_stats<<<64, 256, 0, stream>>>(hbuf, stats, Nn);
  {
    long t = (long)Nn * 64;
    k_bn_apply<<<(unsigned)((t + 255) / 256), 256, 0, stream>>>(hbuf, stats, bn_g, bn_b, sA, Nn);
  }
  gemm(sA, pw_fe2, fe_b2, Nn, 64, 128, /*relu*/2, vh, vf, nullptr);

  // ---- edge geometry ----
  k_edge_geom<<<(unsigned)(((long)E + 255) / 256), 256, 0, stream>>>(pos, ei, dist, cosC, E);

  // ---- interaction layers ----
  for (int l = 0; l < L; ++l) {
    gemm(vh, pw_lin[l], nullptr, Nn, 128, 128, 0, sA, nullptr, nullptr);
    zero(agg, (long)Nn * 128);
    k_edge<<<(unsigned)(((long)E + 127) / 128), 256, 0, stream>>>(
        dist, cosC, ei, ei + E,
        pw_m1[l], mlp_b1 + (size_t)l * 128,
        pw_m2[l], mlp_b2 + (size_t)l * 128,
        sA, agg, E);
    {
      long t = (long)Nn * 128;
      k_pad16<<<(unsigned)((t + 255) / 256), 256, 0, stream>>>(agg, sA, Nn, 128, 128);
    }
    gemm(sA, pw_v1[l], v1_b + (size_t)l * 128, Nn, 128, 128, /*ssp*/1, sB, nullptr, nullptr);
    gemm(sB, pw_v2[l], v2_b + (size_t)l * 128, Nn, 128, 128, 0, vh, vf, vf);
  }

  // ---- readout ----
  gemm(vh, pw_u1, u1_b, Nn, 128, 64, /*ssp*/1, sA, nullptr, nullptr);
  float* s2 = agg;  // reuse as [Nn,32]
  gemm(sA, pw_u2, u2_b, Nn, 64, 32, 0, nullptr, s2, nullptr);
  zero(out, (long)out_size);
  k_scatter_out<<<(unsigned)(((long)Nn * 32 + 255) / 256), 256, 0, stream>>>(s2, batch, out, Nn);
}
